// MistralAttention_69432441307495
// MI455X (gfx1250) — matrix operand order repaired and checked by reading, not compiled, not run on hardware
//
#include <hip/hip_runtime.h>
#include <hip/hip_bf16.h>
#include <math.h>

#define S_LEN 2048
#define EMB   4096
#define NH    32
#define NKV   8
#define DH    128
#define WIN   1024

typedef __bf16 bf16;
typedef __attribute__((ext_vector_type(16))) __bf16 v16bf;
typedef __attribute__((ext_vector_type(8)))  float  v8f;
typedef int v4i __attribute__((vector_size(16)));   // matches builtin param pointee

// ---- CDNA5 async global->LDS path (GLOBAL_LOAD_ASYNC_TO_LDS_B128, ASYNCcnt) ----
#if defined(__has_builtin)
#  if __has_builtin(__builtin_amdgcn_global_load_async_to_lds_b128) && \
      __has_builtin(__builtin_amdgcn_s_wait_asynccnt)
#    define USE_ASYNC_LDS 1
#  endif
#endif
#ifndef USE_ASYNC_LDS
#  define USE_ASYNC_LDS 0
#endif

typedef __attribute__((address_space(1))) v4i* g_v4i_p;
typedef __attribute__((address_space(3))) v4i* l_v4i_p;

__device__ __forceinline__ void cp16(const bf16* g, bf16* l) {
#if USE_ASYNC_LDS
  __builtin_amdgcn_global_load_async_to_lds_b128((g_v4i_p)g, (l_v4i_p)l, 0, 0);
#else
  *(uint4*)l = *(const uint4*)g;
#endif
}
__device__ __forceinline__ void async_wait_all() {
#if USE_ASYNC_LDS
  __builtin_amdgcn_s_wait_asynccnt(0);
#endif
}

// --- WMMA fragment coordinate maps (CDNA5 ISA 7.12.2, 16-bit operands) ---
// A (16x32, MxK): lane holds row M=lane&15; K interleaved by half-wave.
__device__ __forceinline__ int a_k_of(int e, int half) {
  return ((e >> 3) << 4) + (((e >> 1) & 3) << 1) + (e & 1) + (half << 3);
}
// B (32x16, KxN): lane holds col N=lane&15; K interleaved by half-wave (same as A).
__device__ __forceinline__ int b_k_of(int e, int half) {
  return ((e >> 3) << 4) + (e & 7) + (half << 3);
}

// ---------------------------------------------------------------- convert
__global__ void f32_to_bf16_k(const float* __restrict__ in,
                              bf16* __restrict__ out, size_t n) {
  size_t i = (size_t)blockIdx.x * blockDim.x + threadIdx.x;
  if (i < n) out[i] = (bf16)in[i];
}

// ---------------------------------------------------------------- RoPE (in place, bf16)
__global__ void rope_inplace_k(bf16* __restrict__ x, int nh) {
  int idx = blockIdx.x * blockDim.x + threadIdx.x;
  int per_s = nh * (DH / 2);
  int s = idx / per_s;
  if (s >= S_LEN) return;
  int rem = idx % per_s;
  int h = rem / (DH / 2);
  int i = rem % (DH / 2);
  float inv = __powf(10000.0f, -(float)i / 64.0f);
  float ang = (float)s * inv;
  float c = __cosf(ang), sn = __sinf(ang);
  bf16* p = x + (size_t)s * nh * DH + (size_t)h * DH;
  float x1 = (float)p[i];
  float x2 = (float)p[i + DH / 2];
  p[i]          = (bf16)(x1 * c - x2 * sn);
  p[i + DH / 2] = (bf16)(x2 * c + x1 * sn);
}

// ---------------------------------------------------------------- GEMM: C[MxN] = A[MxK] * W[NxK]^T
// 256 threads = 8 waves; block tile 128x128; wave tile 32x64 (2x4 WMMA tiles);
// K-step 32; double-buffered LDS with async global->LDS staging.
#define BM 128
#define BN 128
#define BK 32

__global__ __launch_bounds__(256)
void gemm_bf16_wmma_k(const bf16* __restrict__ A, const bf16* __restrict__ W,
                      float* __restrict__ Cf, bf16* __restrict__ Cb,
                      int M, int N, int K) {
  __shared__ __align__(16) bf16 As[2][BM][BK + 8];
  __shared__ __align__(16) bf16 Bs[2][BN][BK + 8];
  const int tid  = threadIdx.x;
  const int lane = tid & 31;
  const int wid  = tid >> 5;
  const int half = lane >> 4;
  const int l16  = lane & 15;
  const int wm   = wid >> 1;   // 0..3  -> 32-row slab
  const int wn   = wid & 1;    // 0..1  -> 64-col slab
  const int m0   = blockIdx.x * BM;
  const int n0   = blockIdx.y * BN;

  v8f c[2][4];
#pragma unroll
  for (int i = 0; i < 2; ++i)
#pragma unroll
    for (int j = 0; j < 4; ++j)
#pragma unroll
      for (int r = 0; r < 8; ++r) c[i][j][r] = 0.0f;

  auto stage = [&](int buf, int k0) {
    // A tile: 128x32 bf16 = 512 uint4 (2/thread); B tile likewise.
#pragma unroll
    for (int i = 0; i < 2; ++i) {
      int idx  = tid + i * 256;
      int row  = idx >> 2;
      int quad = idx & 3;
      cp16(A + (size_t)(m0 + row) * K + k0 + quad * 8, &As[buf][row][quad * 8]);
    }
#pragma unroll
    for (int i = 0; i < 2; ++i) {
      int idx  = tid + i * 256;
      int row  = idx >> 2;
      int quad = idx & 3;
      cp16(W + (size_t)(n0 + row) * K + k0 + quad * 8, &Bs[buf][row][quad * 8]);
    }
  };

  stage(0, 0);
  int cur = 0;
  for (int k0 = 0; k0 < K; k0 += BK) {
    async_wait_all();
    __syncthreads();   // tile[cur] visible; also WAR-protects tile[cur^1]
    if (k0 + BK < K) stage(cur ^ 1, k0 + BK);

    v16bf a[2], b[4];
#pragma unroll
    for (int mt = 0; mt < 2; ++mt)
#pragma unroll
      for (int e = 0; e < 16; ++e)
        a[mt][e] = As[cur][wm * 32 + mt * 16 + l16][a_k_of(e, half)];
#pragma unroll
    for (int nt = 0; nt < 4; ++nt)
#pragma unroll
      for (int e = 0; e < 16; ++e)
        b[nt][e] = Bs[cur][wn * 64 + nt * 16 + l16][b_k_of(e, half)];

#pragma unroll
    for (int mt = 0; mt < 2; ++mt)
#pragma unroll
      for (int nt = 0; nt < 4; ++nt)
        c[mt][nt] = __builtin_amdgcn_wmma_f32_16x16x32_bf16(
            false, a[mt], false, b[nt], (short)0, c[mt][nt], false, false);
    cur ^= 1;
  }

#pragma unroll
  for (int mt = 0; mt < 2; ++mt)
#pragma unroll
    for (int nt = 0; nt < 4; ++nt)
#pragma unroll
      for (int r = 0; r < 8; ++r) {
        int row = m0 + wm * 32 + mt * 16 + r + (half << 3);
        int col = n0 + wn * 64 + nt * 16 + l16;
        float v = c[mt][nt][r];
        if (Cb) Cb[(size_t)row * N + col] = (bf16)v;
        else    Cf[(size_t)row * N + col] = v;
      }
}

// ---------------------------------------------------------------- banded flash attention
// One wave per (16-query tile, head). Streams 32-key blocks over the band
// |i-j| >= WIN (reference masks |i-j| < WIN to -inf). Online softmax.
__global__ __launch_bounds__(32)
void attn_band_k(const bf16* __restrict__ Q, const bf16* __restrict__ Kc,
                 const bf16* __restrict__ Vc, bf16* __restrict__ O) {
  __shared__ __align__(16) bf16 Qs[16][DH];
  __shared__ __align__(16) bf16 Ks[32][DH];
  __shared__ __align__(16) bf16 Vs[32][DH];
  __shared__ __align__(16) bf16 Ps[16][32];

  const int lane = threadIdx.x;
  const int half = lane >> 4;
  const int l16  = lane & 15;
  const int q0   = blockIdx.x * 16;
  const int h    = blockIdx.y;
  const int kvh  = h / (NH / NKV);
  const float scale = 0.08838834764831845f;  // 1/sqrt(128)

  // stage Q tile (16x128)
#pragma unroll
  for (int i = 0; i < 8; ++i) {
    int idx  = lane + i * 32;   // 0..255
    int row  = idx >> 4;
    int quad = idx & 15;
    cp16(Q + (size_t)(q0 + row) * (NH * DH) + (size_t)h * DH + quad * 8,
         &Qs[row][quad * 8]);
  }
  async_wait_all();
  __syncthreads();

  v16bf qf[4];
#pragma unroll
  for (int cch = 0; cch < 4; ++cch)
#pragma unroll
    for (int e = 0; e < 16; ++e)
      qf[cch][e] = Qs[l16][cch * 32 + a_k_of(e, half)];

  v8f o[8];
  float mrow[8], lrow[8];
#pragma unroll
  for (int g = 0; g < 8; ++g)
#pragma unroll
    for (int r = 0; r < 8; ++r) o[g][r] = 0.0f;
#pragma unroll
  for (int r = 0; r < 8; ++r) { mrow[r] = -1e30f; lrow[r] = 0.0f; }

  for (int side = 0; side < 2; ++side) {
    int jbeg, jend;
    if (side == 0) {                       // j >= i + WIN
      if (q0 + WIN >= S_LEN) continue;
      jbeg = (q0 + WIN) & ~31;
      jend = S_LEN;
    } else {                               // j <= i - WIN
      if (q0 + 15 < WIN) continue;
      jbeg = 0;
      jend = (((q0 + 15 - WIN) >> 5) << 5) + 32;
    }

    for (int j0 = jbeg; j0 < jend; j0 += 32) {
      // stage K,V blocks (each 32x128)
#pragma unroll
      for (int i = 0; i < 16; ++i) {
        int idx  = lane + i * 32;   // 0..511
        int row  = idx >> 4;
        int quad = idx & 15;
        cp16(Kc + (size_t)(j0 + row) * (NKV * DH) + (size_t)kvh * DH + quad * 8,
             &Ks[row][quad * 8]);
        cp16(Vc + (size_t)(j0 + row) * (NKV * DH) + (size_t)kvh * DH + quad * 8,
             &Vs[row][quad * 8]);
      }
      async_wait_all();
      __syncthreads();

      // scores: two 16x16 tiles, K-dim 128 = 4 WMMA each
      v8f sc[2];
#pragma unroll
      for (int t = 0; t < 2; ++t) {
        v8f acc;
#pragma unroll
        for (int r = 0; r < 8; ++r) acc[r] = 0.0f;
#pragma unroll
        for (int cch = 0; cch < 4; ++cch) {
          v16bf kf;
#pragma unroll
          for (int e = 0; e < 16; ++e)
            kf[e] = Ks[t * 16 + l16][cch * 32 + b_k_of(e, half)];
          acc = __builtin_amdgcn_wmma_f32_16x16x32_bf16(
              false, qf[cch], false, kf, (short)0, acc, false, false);
        }
        sc[t] = acc;
      }

      // mask + scale, row max (16-lane half-wave reduction)
      float mx[8];
#pragma unroll
      for (int r = 0; r < 8; ++r) {
        int irow = q0 + r + (half << 3);
        int jc0 = j0 + l16, jc1 = j0 + 16 + l16;
        int d0 = irow - jc0; if (d0 < 0) d0 = -d0;
        int d1 = irow - jc1; if (d1 < 0) d1 = -d1;
        float v0 = (d0 >= WIN) ? sc[0][r] * scale : -1e30f;
        float v1 = (d1 >= WIN) ? sc[1][r] * scale : -1e30f;
        sc[0][r] = v0; sc[1][r] = v1;
        mx[r] = fmaxf(v0, v1);
      }
#pragma unroll
      for (int off = 1; off < 16; off <<= 1)
#pragma unroll
        for (int r = 0; r < 8; ++r)
          mx[r] = fmaxf(mx[r], __shfl_xor(mx[r], off, 32));

      // online softmax update
      float alpha[8], rs[8];
#pragma unroll
      for (int r = 0; r < 8; ++r) {
        float mnew = fmaxf(mrow[r], mx[r]);
        alpha[r] = __expf(mrow[r] - mnew);
        mrow[r] = mnew;
        float p0 = (sc[0][r] > -5e29f) ? __expf(sc[0][r] - mnew) : 0.0f;
        float p1 = (sc[1][r] > -5e29f) ? __expf(sc[1][r] - mnew) : 0.0f;
        sc[0][r] = p0; sc[1][r] = p1;
        rs[r] = p0 + p1;
      }
#pragma unroll
      for (int off = 1; off < 16; off <<= 1)
#pragma unroll
        for (int r = 0; r < 8; ++r)
          rs[r] += __shfl_xor(rs[r], off, 32);
#pragma unroll
      for (int r = 0; r < 8; ++r)
        lrow[r] = lrow[r] * alpha[r] + rs[r];
#pragma unroll
      for (int g = 0; g < 8; ++g)
#pragma unroll
        for (int r = 0; r < 8; ++r)
          o[g][r] *= alpha[r];

      // re-stage P (C layout -> A layout via LDS)
#pragma unroll
      for (int r = 0; r < 8; ++r) {
        Ps[r + (half << 3)][l16]      = (bf16)sc[0][r];
        Ps[r + (half << 3)][16 + l16] = (bf16)sc[1][r];
      }
      __syncthreads();

      v16bf pf;
#pragma unroll
      for (int e = 0; e < 16; ++e)
        pf[e] = Ps[l16][a_k_of(e, half)];
#pragma unroll
      for (int g = 0; g < 8; ++g) {
        v16bf vf;
#pragma unroll
        for (int e = 0; e < 16; ++e)
          vf[e] = Vs[b_k_of(e, half)][g * 16 + l16];
        o[g] = __builtin_amdgcn_wmma_f32_16x16x32_bf16(
            false, pf, false, vf, (short)0, o[g], false, false);
      }
      __syncthreads();
    }
  }

  // normalize and store (bf16, [S, H*D] layout)
#pragma unroll
  for (int g = 0; g < 8; ++g)
#pragma unroll
    for (int r = 0; r < 8; ++r) {
      int row = q0 + r + (half << 3);
      float denom = (lrow[r] > 0.0f) ? lrow[r] : 1.0f;
      O[(size_t)row * (NH * DH) + (size_t)h * DH + g * 16 + l16] =
          (bf16)(o[g][r] / denom);
    }
}

// ---------------------------------------------------------------- launcher
extern "C" void kernel_launch(void* const* d_in, const int* in_sizes, int n_in,
                              void* d_out, int out_size, void* d_ws, size_t ws_size,
                              hipStream_t stream) {
  const float* hs = (const float*)d_in[0];
  const float* Wq = (const float*)d_in[1];
  const float* Wk = (const float*)d_in[2];
  const float* Wv = (const float*)d_in[3];
  const float* Wo = (const float*)d_in[4];
  float* out = (float*)d_out;
  char* ws = (char*)d_ws;

  const size_t SZ_X  = (size_t)S_LEN * EMB * 2;        // 16 MB
  const size_t SZ_WQ = (size_t)EMB * EMB * 2;          // 32 MB
  const size_t SZ_WK = (size_t)(NKV * DH) * EMB * 2;   //  8 MB
  const size_t SZ_K  = (size_t)S_LEN * NKV * DH * 2;   //  4 MB

  bf16* Xb  = (bf16*)(ws);
  bf16* Wqb = (bf16*)(ws + SZ_X);
  bf16* Wkb = (bf16*)(ws + SZ_X + SZ_WQ);
  bf16* Wvb = (bf16*)(ws + SZ_X + SZ_WQ + SZ_WK);
  bf16* Qb  = (bf16*)(ws + SZ_X + SZ_WQ + 2 * SZ_WK);
  bf16* Kb  = (bf16*)(ws + SZ_X + SZ_WQ + 2 * SZ_WK + SZ_X);
  bf16* Vb  = (bf16*)(ws + SZ_X + SZ_WQ + 2 * SZ_WK + SZ_X + SZ_K);
  bf16* AOb = Xb;   // X no longer needed after QKV projections
  bf16* Wob = Wqb;  // Wq no longer needed after Q projection

  auto cvt = [&](const float* src, bf16* dst, size_t n) {
    f32_to_bf16_k<<<dim3((unsigned)((n + 255) / 256)), 256, 0, stream>>>(src, dst, n);
  };

  cvt(hs, Xb, (size_t)S_LEN * EMB);
  cvt(Wq, Wqb, (size_t)EMB * EMB);
  cvt(Wk, Wkb, (size_t)(NKV * DH) * EMB);
  cvt(Wv, Wvb, (size_t)(NKV * DH) * EMB);

  gemm_bf16_wmma_k<<<dim3(S_LEN / BM, (NH * DH) / BN), 256, 0, stream>>>(
      Xb, Wqb, nullptr, Qb, S_LEN, NH * DH, EMB);
  gemm_bf16_wmma_k<<<dim3(S_LEN / BM, (NKV * DH) / BN), 256, 0, stream>>>(
      Xb, Wkb, nullptr, Kb, S_LEN, NKV * DH, EMB);
  gemm_bf16_wmma_k<<<dim3(S_LEN / BM, (NKV * DH) / BN), 256, 0, stream>>>(
      Xb, Wvb, nullptr, Vb, S_LEN, NKV * DH, EMB);

  {
    size_t nq = (size_t)S_LEN * NH * (DH / 2);
    rope_inplace_k<<<(unsigned)((nq + 255) / 256), 256, 0, stream>>>(Qb, NH);
    size_t nk = (size_t)S_LEN * NKV * (DH / 2);
    rope_inplace_k<<<(unsigned)((nk + 255) / 256), 256, 0, stream>>>(Kb, NKV);
  }

  attn_band_k<<<dim3(S_LEN / 16, NH), 32, 0, stream>>>(Qb, Kb, Vb, AOb);

  cvt(Wo, Wob, (size_t)EMB * EMB);
  gemm_bf16_wmma_k<<<dim3(S_LEN / BM, EMB / BN), 256, 0, stream>>>(
      AOb, Wob, out, nullptr, S_LEN, EMB, EMB);
}
